// RetNetPoST_42125039239462
// MI455X (gfx1250) — matrix operand order repaired and checked by reading, not compiled, not run on hardware
//
#include <hip/hip_runtime.h>

// ---------------------------------------------------------------------------
// CDNA5 (gfx1250) GLA block: bf16 WMMA everywhere, wave32.
// bf16-preconverted GEMM operands, double-buffered LDS, vector fragments,
// TDM (tensor_load_to_lds) async staging of GEMM A-tiles.
// ---------------------------------------------------------------------------

typedef __attribute__((ext_vector_type(16))) __bf16 v16bf;
typedef __attribute__((ext_vector_type(8)))  __bf16 v8bf;
typedef __attribute__((ext_vector_type(8)))  float  v8f;
typedef __attribute__((ext_vector_type(4)))  unsigned int u32x4;
typedef __attribute__((ext_vector_type(4)))  int i32x4;
typedef __attribute__((ext_vector_type(8)))  int i32x8;

#define D_MODEL 1024
#define HEADS   8
#define DK      128
#define DV      128
#define SEQ_T   2048
#define BATCH   4
#define CHUNK   64
#define LOG_T_REF 6.238324625039508f   // log(512)

#if __has_builtin(__builtin_amdgcn_tensor_load_to_lds) && \
    __has_builtin(__builtin_amdgcn_s_wait_tensorcnt)
#define HAVE_TDM 1
#else
#define HAVE_TDM 0
#endif

__device__ __forceinline__ v8f wmma_bf16(v16bf a, v16bf b, v8f c) {
    return __builtin_amdgcn_wmma_f32_16x16x32_bf16(
        false, a, false, b, (short)0, c, false, false);
}

__device__ __forceinline__ v16bf cat16(v8bf lo, v8bf hi) {
    return __builtin_shufflevector(lo, hi, 0, 1, 2, 3, 4, 5, 6, 7,
                                           8, 9, 10, 11, 12, 13, 14, 15);
}

// A-matrix fragment (16x32 bf16), logical A[m][k] at p[m*ld + k], ld % 8 == 0.
// ISA layout: k = (v>>2)*16 + h*8 + (v&3)*2 -> runs [k0+h*8,+8) and [k0+16+h*8,+8)
__device__ __forceinline__ v16bf frag_a_v(const __bf16* p, int ld, int row0,
                                          int k0, int lane) {
    const int half = (lane >> 4) & 1, r = lane & 15;
    const __bf16* base = p + (size_t)(row0 + r) * ld + k0 + half * 8;
    return cat16(*reinterpret_cast<const v8bf*>(base),
                 *reinterpret_cast<const v8bf*>(base + 16));
}

// B-matrix fragment (32x16 bf16), logical B[k][n] stored TRANSPOSED pT[n*ld+k].
// ISA layout: k = (v>>2)*16 + h*8 + (v&3)*2 -> runs [k0+h*8,+8) and [k0+16+h*8,+8)
__device__ __forceinline__ v16bf frag_b_v(const __bf16* pT, int ld, int col0,
                                          int k0, int lane) {
    const int half = (lane >> 4) & 1, r = lane & 15;
    const __bf16* base = pT + (size_t)(col0 + r) * ld + k0 + half * 8;
    return cat16(*reinterpret_cast<const v8bf*>(base),
                 *reinterpret_cast<const v8bf*>(base + 16));
}

#if HAVE_TDM
// TDM: async-DMA a [128 rows x 32 cols] bf16 tile (row stride K elems) from
// global into LDS at lds_addr, inserting 16B pad after every 64B row so the
// LDS layout matches As[128][40]. D# per CDNA5 ISA ch.8.
__device__ __forceinline__ void tdm_load_tile_a(const __bf16* gsrc,
                                                unsigned lds_addr,
                                                int K, int M) {
    const unsigned long long ga = (unsigned long long)(size_t)gsrc;
    u32x4 g0;
    g0[0] = 1u;                                       // count=1, user mode
    g0[1] = lds_addr;                                 // lds_addr[31:0]
    g0[2] = (unsigned)(ga & 0xffffffffull);           // global_addr[31:0]
    g0[3] = (unsigned)((ga >> 32) & 0x01ffffffull)    // global_addr[56:32]
            | (2u << 30);                             // type=2 (image)
    i32x8 g1;
    g1[0] = (1 << 16)        // data_size = 2 bytes
          | (1 << 20)        // pad_enable
          | (3 << 22)        // pad_interval: 16 DWORDs (64B)
          | (3 << 25);       // pad_amount: 4 DWORDs (16B)
    g1[1] = (K & 0xffff) << 16;                       // tensor_dim0[15:0]
    g1[2] = ((K >> 16) & 0xffff) | ((M & 0xffff) << 16);  // dim0 hi | dim1 lo
    g1[3] = ((M >> 16) & 0xffff) | (32 << 16);        // dim1 hi | tile_dim0=32
    g1[4] = 128;                                      // tile_dim1=128, dim2=0
    g1[5] = K;                                        // tensor_dim0_stride lo
    g1[6] = 0;                                        // stride hi / dim1_stride
    g1[7] = 0;
    const i32x4 z4 = {0, 0, 0, 0};
#if defined(__clang_major__) && __clang_major__ >= 23
    const i32x8 z8 = {0, 0, 0, 0, 0, 0, 0, 0};
    __builtin_amdgcn_tensor_load_to_lds(g0, g1, z4, z4, z8, 0);
#else
    __builtin_amdgcn_tensor_load_to_lds(g0, g1, z4, z4, 0);
#endif
}
#endif

// ---------------------------------------------------------------------------
// fp32 -> bf16 bulk conversion (8 elems / thread)
// ---------------------------------------------------------------------------
__global__ __launch_bounds__(256)
void to_bf16_kernel(const float* __restrict__ x, __bf16* __restrict__ y, long n8) {
    const long i = (long)blockIdx.x * 256 + threadIdx.x;
    if (i >= n8) return;
    const float4 a = reinterpret_cast<const float4*>(x)[2 * i];
    const float4 b = reinterpret_cast<const float4*>(x)[2 * i + 1];
    v8bf o;
    o[0] = (__bf16)a.x; o[1] = (__bf16)a.y; o[2] = (__bf16)a.z; o[3] = (__bf16)a.w;
    o[4] = (__bf16)b.x; o[5] = (__bf16)b.y; o[6] = (__bf16)b.z; o[7] = (__bf16)b.w;
    reinterpret_cast<v8bf*>(y)[i] = o;
}

// ---------------------------------------------------------------------------
// GEMM: Y[M,N] = X[M,K] @ W[K,N], bf16 in, fp32 out.
// Block tile 128(M) x 128(N); 8 waves = 4(M) x 2(N); wave = 2x4 WMMA tiles.
// Double-buffered LDS; A staged via TDM; B transposed [n][k] ld 40.
// ---------------------------------------------------------------------------
__global__ __launch_bounds__(256)
void gemm_bf16(const __bf16* __restrict__ X, const __bf16* __restrict__ W,
               float* __restrict__ Y, int M, int N, int K) {
    __shared__ alignas(16) __bf16 As[2][128][40];
    __shared__ alignas(16) __bf16 BsT[2][128][40];

    const int tid = threadIdx.x;
    const int lane = tid & 31, wave = tid >> 5;
    const int wm = wave & 3, wn = wave >> 2;
    const int m0 = blockIdx.y * 128, n0 = blockIdx.x * 128;
    const int half = (lane >> 4) & 1, r = lane & 15;
    const int KT = K / 32;

    v8f acc[2][4] = {};
    v8bf rb[2];
#if !HAVE_TDM
    v8bf ra[2];
#endif

#if HAVE_TDM
    const unsigned lds_a[2] = {(unsigned)(size_t)&As[0][0][0],
                               (unsigned)(size_t)&As[1][0][0]};
#else
    auto load_a_regs = [&](int k0) {
#pragma unroll
        for (int i = 0; i < 2; ++i) {
            const int e = i * 256 + tid;
            ra[i] = *reinterpret_cast<const v8bf*>(
                &X[(size_t)(m0 + (e >> 2)) * K + k0 + (e & 3) * 8]);
        }
    };
    auto store_a_lds = [&](int buf) {
#pragma unroll
        for (int i = 0; i < 2; ++i) {
            const int e = i * 256 + tid;
            *reinterpret_cast<v8bf*>(&As[buf][e >> 2][(e & 3) * 8]) = ra[i];
        }
    };
#endif
    auto load_b_regs = [&](int k0) {
#pragma unroll
        for (int i = 0; i < 2; ++i) {
            const int e = i * 256 + tid;
            rb[i] = *reinterpret_cast<const v8bf*>(
                &W[(size_t)(k0 + (e >> 4)) * N + n0 + (e & 15) * 8]);
        }
    };
    auto store_b_lds = [&](int buf) {
#pragma unroll
        for (int i = 0; i < 2; ++i) {
            const int e = i * 256 + tid;
            const int kr = e >> 4, nc = (e & 15) * 8;
#pragma unroll
            for (int j = 0; j < 8; ++j) BsT[buf][nc + j][kr] = rb[i][j];
        }
    };

    // prologue: stage tile 0
#if HAVE_TDM
    if (wave == 0)
        tdm_load_tile_a(&X[(size_t)m0 * K], lds_a[0], K, M);
#else
    load_a_regs(0);
#endif
    load_b_regs(0);
#if !HAVE_TDM
    store_a_lds(0);
#endif
    store_b_lds(0);
#if HAVE_TDM
    __builtin_amdgcn_s_wait_tensorcnt(0);
#endif

#pragma unroll 1
    for (int kt = 0; kt < KT; ++kt) {
        const int cur = kt & 1;
        __syncthreads();
        if (kt + 1 < KT) {
#if HAVE_TDM
            if (wave == 0)
                tdm_load_tile_a(&X[(size_t)m0 * K + (kt + 1) * 32],
                                lds_a[cur ^ 1], K, M);
#else
            load_a_regs((kt + 1) * 32);
#endif
            load_b_regs((kt + 1) * 32);
        }

        v16bf bfr[4];
#pragma unroll
        for (int ni = 0; ni < 4; ++ni)
            bfr[ni] = frag_b_v(&BsT[cur][0][0], 40, wn * 64 + ni * 16, 0, lane);
#pragma unroll
        for (int mi = 0; mi < 2; ++mi) {
            v16bf af = frag_a_v(&As[cur][0][0], 40, wm * 32 + mi * 16, 0, lane);
#pragma unroll
            for (int ni = 0; ni < 4; ++ni)
                acc[mi][ni] = wmma_bf16(af, bfr[ni], acc[mi][ni]);
        }

        if (kt + 1 < KT) {
#if !HAVE_TDM
            store_a_lds(cur ^ 1);
#endif
            store_b_lds(cur ^ 1);
        }
#if HAVE_TDM
        __builtin_amdgcn_s_wait_tensorcnt(0);   // next A-tile landed in LDS
#endif
    }

    // C/D layout: VGPR j -> row = half*8 + j, col = r
#pragma unroll
    for (int mi = 0; mi < 2; ++mi)
#pragma unroll
        for (int ni = 0; ni < 4; ++ni)
#pragma unroll
            for (int j = 0; j < 8; ++j) {
                const int row = m0 + wm * 32 + mi * 16 + half * 8 + j;
                const int col = n0 + wn * 64 + ni * 16 + r;
                Y[(size_t)row * N + col] = acc[mi][ni][j];
            }
}

// ---------------------------------------------------------------------------
// Depthwise causal conv (width 4) + SiLU, 4 channels per thread.
// ---------------------------------------------------------------------------
__global__ __launch_bounds__(256)
void conv_silu_kernel(const float* __restrict__ x, const float* __restrict__ w,
                      float* __restrict__ y, int T, int total4) {
    const int idx4 = blockIdx.x * 256 + threadIdx.x;
    if (idx4 >= total4) return;
    const long idx = (long)idx4 * 4;
    const int c = (int)(idx & (D_MODEL - 1));
    const int t = (int)((idx >> 10) % T);
    float4 acc = make_float4(0.f, 0.f, 0.f, 0.f);
#pragma unroll
    for (int i = 0; i < 4; ++i) {
        if (t - 3 + i >= 0) {
            const float4 xv = *reinterpret_cast<const float4*>(
                &x[idx + (long)(i - 3) * D_MODEL]);
            acc.x += xv.x * w[(c + 0) * 4 + i];
            acc.y += xv.y * w[(c + 1) * 4 + i];
            acc.z += xv.z * w[(c + 2) * 4 + i];
            acc.w += xv.w * w[(c + 3) * 4 + i];
        }
    }
    float4 o;
    o.x = acc.x / (1.f + __expf(-acc.x));
    o.y = acc.y / (1.f + __expf(-acc.y));
    o.z = acc.z / (1.f + __expf(-acc.z));
    o.w = acc.w / (1.f + __expf(-acc.w));
    *reinterpret_cast<float4*>(&y[idx]) = o;
}

// ---------------------------------------------------------------------------
// Per-head constants: hc[0..7] = exp(A_log[h]); hc[8..15] = alpha[h]
// ---------------------------------------------------------------------------
__global__ void headconst_kernel(const float* __restrict__ A_log_base,
                                 const float* __restrict__ A_log_delta,
                                 float* __restrict__ hc) {
    if (threadIdx.x != 0 || blockIdx.x != 0) return;
    float alog[HEADS];
    const float base = A_log_base[0];
    alog[0] = base;
    float cum = 0.f;
    for (int h = 1; h < HEADS; ++h) {
        const float x = A_log_delta[h - 1];
        cum += (x > 20.f) ? x : log1pf(__expf(x));
        alog[h] = base + cum;
    }
    const float mean_gap = (alog[HEADS - 1] - alog[0]) / (float)(HEADS - 1);
    for (int h = 0; h < HEADS; ++h) {
        const float cu = alog[h] - alog[0];
        float a = (float)(HEADS - 1 - h) / (float)(HEADS - 1) +
                  (cu - (float)h * mean_gap) / LOG_T_REF;
        a = fminf(fmaxf(a, 0.f), 1.f);
        hc[h] = __expf(alog[h]);
        hc[HEADS + h] = a;
    }
}

// ---------------------------------------------------------------------------
// gk = -exp(A_log[h]) * softplus(a + dt_bias) * (t+1)^(-alpha[h])  (in place)
// ---------------------------------------------------------------------------
__global__ __launch_bounds__(256)
void gk_kernel(float* __restrict__ a_buf, const float* __restrict__ dt_bias,
               const float* __restrict__ hc, int total4, int T) {
    const int idx4 = blockIdx.x * 256 + threadIdx.x;
    if (idx4 >= total4) return;
    const long idx = (long)idx4 * 4;
    const int cd = (int)(idx & (D_MODEL - 1));
    const int t = (int)((idx >> 10) % T);
    const int h = cd >> 7;
    const float4 a = *reinterpret_cast<const float4*>(&a_buf[idx]);
    const float4 db = *reinterpret_cast<const float4*>(&dt_bias[cd]);
    const float ps = -hc[h] * __expf(-hc[HEADS + h] * __logf((float)(t + 1)));
    float4 o;
    float x;
    x = a.x + db.x; o.x = ps * ((x > 20.f) ? x : log1pf(__expf(x)));
    x = a.y + db.y; o.y = ps * ((x > 20.f) ? x : log1pf(__expf(x)));
    x = a.z + db.z; o.z = ps * ((x > 20.f) ? x : log1pf(__expf(x)));
    x = a.w + db.w; o.w = ps * ((x > 20.f) ? x : log1pf(__expf(x)));
    *reinterpret_cast<float4*>(&a_buf[idx]) = o;
}

// ---------------------------------------------------------------------------
// Chunked GLA scan. One workgroup per (b,h). 8 waves.
// State (128x128 f32) in registers; wave w owns rows [w*16, w*16+16).
// ---------------------------------------------------------------------------
__global__ __launch_bounds__(256)
void gla_kernel(const float* __restrict__ q, const float* __restrict__ k,
                const float* __restrict__ v, const float* __restrict__ gk,
                float* __restrict__ o, int B, int T) {
    __shared__ alignas(16) __bf16 st_t[DV][DK + 8];      // state^T: [e][d]
    __shared__ alignas(16) __bf16 qs_s[CHUNK][DK + 8];   // [i][d]
    __shared__ alignas(16) __bf16 ki_s[CHUNK][DK + 8];   // [j][d] (= (ki^T)^T)
    __shared__ alignas(16) __bf16 ki_t[DK][CHUNK + 8];   // [d][j]
    __shared__ alignas(16) __bf16 vv_t[DV][CHUNK + 8];   // v^T: [e][j]
    __shared__ alignas(16) __bf16 A_s[CHUNK][CHUNK + 8]; // [i][j]
    __shared__ float eg_s[DK];                           // exp(gtot[d])

    const int tid = threadIdx.x, lane = tid & 31, wave = tid >> 5;
    const int half = (lane >> 4) & 1, r = lane & 15;
    const int b = blockIdx.x >> 3, h = blockIdx.x & 7;
    const float scale = 0.08838834764831845f;  // 128^-0.5

    const size_t bo = (size_t)b * T * D_MODEL + (size_t)h * DK;
    const float* qp = q + bo;
    const float* kp = k + bo;
    const float* vp = v + bo;
    const float* gp = gk + bo;
    float* op = o + bo;

    v8f st[8] = {};

#pragma unroll 1
    for (int c = 0; c < T / CHUNK; ++c) {
        const size_t tok0 = (size_t)c * CHUNK;

        // ---- P0: spill state to bf16 LDS (transposed), vector stores
#pragma unroll
        for (int n = 0; n < 8; ++n) {
            v8bf sb;
#pragma unroll
            for (int j = 0; j < 8; ++j) sb[j] = (__bf16)st[n][j];
            *reinterpret_cast<v8bf*>(
                &st_t[n * 16 + r][wave * 16 + half * 8]) = sb;
        }

        // ---- P1: gate cumsum + staging (loads batched to overlap latency)
        if (tid < DK) {
            const int d = tid;
            float gc = 0.f;
#pragma unroll 1
            for (int i8 = 0; i8 < CHUNK / 8; ++i8) {
                float gv[8], qv[8], kv[8];
#pragma unroll
                for (int j = 0; j < 8; ++j) {
                    const size_t off = (tok0 + i8 * 8 + j) * D_MODEL + d;
                    gv[j] = gp[off];
                    qv[j] = qp[off];
                    kv[j] = kp[off];
                }
                v8bf kib;
#pragma unroll
                for (int j = 0; j < 8; ++j) {
                    const int i = i8 * 8 + j;
                    gc += gv[j];
                    qs_s[i][d] = (__bf16)(qv[j] * scale * __expf(gc));
                    const __bf16 kb = (__bf16)(kv[j] * __expf(-gc));
                    ki_s[i][d] = kb;
                    kib[j] = kb;
                }
                *reinterpret_cast<v8bf*>(&ki_t[d][i8 * 8]) = kib;
            }
            eg_s[d] = __expf(gc);
        } else {
            const int e = tid - DK;
#pragma unroll 1
            for (int i8 = 0; i8 < CHUNK / 8; ++i8) {
                float vv[8];
#pragma unroll
                for (int j = 0; j < 8; ++j)
                    vv[j] = vp[(tok0 + i8 * 8 + j) * D_MODEL + e];
                v8bf vb;
#pragma unroll
                for (int j = 0; j < 8; ++j) vb[j] = (__bf16)vv[j];
                *reinterpret_cast<v8bf*>(&vv_t[e][i8 * 8]) = vb;
            }
        }
        __syncthreads();

        // ---- P2: A = qs @ ki^T, causal mask. 16 tiles, 2 per wave.
#pragma unroll
        for (int t2 = 0; t2 < 2; ++t2) {
            const int tile = wave * 2 + t2;
            const int mi = tile >> 2, ni = tile & 3;
            v8f acc = {};
#pragma unroll
            for (int ks = 0; ks < 4; ++ks) {
                v16bf af = frag_a_v(&qs_s[0][0], DK + 8, mi * 16, ks * 32, lane);
                v16bf bf = frag_b_v(&ki_s[0][0], DK + 8, ni * 16, ks * 32, lane);
                acc = wmma_bf16(af, bf, acc);
            }
#pragma unroll
            for (int j = 0; j < 8; ++j) {
                const int row = mi * 16 + half * 8 + j, col = ni * 16 + r;
                A_s[row][col] = (__bf16)((row >= col) ? acc[j] : 0.f);
            }
        }
        __syncthreads();

        // ---- P3: o = A @ v + qs @ state. 32 tiles, 4 per wave.
#pragma unroll
        for (int t4 = 0; t4 < 4; ++t4) {
            const int tile = wave * 4 + t4;
            const int mi = tile >> 3, ni = tile & 7;
            v8f acc = {};
#pragma unroll
            for (int ks = 0; ks < 2; ++ks) {
                v16bf af = frag_a_v(&A_s[0][0], CHUNK + 8, mi * 16, ks * 32, lane);
                v16bf bf = frag_b_v(&vv_t[0][0], CHUNK + 8, ni * 16, ks * 32, lane);
                acc = wmma_bf16(af, bf, acc);
            }
#pragma unroll
            for (int ks = 0; ks < 4; ++ks) {
                v16bf af = frag_a_v(&qs_s[0][0], DK + 8, mi * 16, ks * 32, lane);
                v16bf bf = frag_b_v(&st_t[0][0], DK + 8, ni * 16, ks * 32, lane);
                acc = wmma_bf16(af, bf, acc);
            }
#pragma unroll
            for (int j = 0; j < 8; ++j) {
                const int row = mi * 16 + half * 8 + j, col = ni * 16 + r;
                op[(tok0 + row) * D_MODEL + col] = acc[j];
            }
        }

        // ---- P4: U = ki^T @ v ; state = e^{gtot[d]} (state + U)
        v16bf afk[2];
#pragma unroll
        for (int ks = 0; ks < 2; ++ks)
            afk[ks] = frag_a_v(&ki_t[0][0], CHUNK + 8, wave * 16, ks * 32, lane);
#pragma unroll
        for (int n = 0; n < 8; ++n) {
            v8f u = {};
#pragma unroll
            for (int ks = 0; ks < 2; ++ks) {
                v16bf bf = frag_b_v(&vv_t[0][0], CHUNK + 8, n * 16, ks * 32, lane);
                u = wmma_bf16(afk[ks], bf, u);
            }
#pragma unroll
            for (int j = 0; j < 8; ++j) {
                const float eg = eg_s[wave * 16 + half * 8 + j];
                st[n][j] = eg * (st[n][j] + u[j]);
            }
        }
        __syncthreads();
    }
}

// ---------------------------------------------------------------------------
// RMS norm over DV + SiLU(g) gate, in place. One wave32 per row of 128.
// ---------------------------------------------------------------------------
__global__ __launch_bounds__(256)
void rmsgate_kernel(float* __restrict__ o, const float* __restrict__ g,
                    const float* __restrict__ w, int rows) {
    const int wave = threadIdx.x >> 5, lane = threadIdx.x & 31;
    const int row = blockIdx.x * 8 + wave;
    if (row >= rows) return;
    float* op = o + (size_t)row * DV;
    const float* gp = g + (size_t)row * DV;
    const int e0 = lane * 4;

    float4 xv = *reinterpret_cast<const float4*>(op + e0);
    float ss = xv.x * xv.x + xv.y * xv.y + xv.z * xv.z + xv.w * xv.w;
#pragma unroll
    for (int m = 16; m >= 1; m >>= 1) ss += __shfl_xor(ss, m, 32);
    const float inv = rsqrtf(ss * (1.f / (float)DV) + 1e-5f);

    float4 gv = *reinterpret_cast<const float4*>(gp + e0);
    float4 wv = *reinterpret_cast<const float4*>(w + e0);
    float4 out;
    out.x = xv.x * inv * wv.x * (gv.x / (1.f + __expf(-gv.x)));
    out.y = xv.y * inv * wv.y * (gv.y / (1.f + __expf(-gv.y)));
    out.z = xv.z * inv * wv.z * (gv.z / (1.f + __expf(-gv.z)));
    out.w = xv.w * inv * wv.w * (gv.w / (1.f + __expf(-gv.w)));
    *reinterpret_cast<float4*>(op + e0) = out;
}

// ---------------------------------------------------------------------------
extern "C" void kernel_launch(void* const* d_in, const int* in_sizes, int n_in,
                              void* d_out, int out_size, void* d_ws, size_t ws_size,
                              hipStream_t stream) {
    const float* hs   = (const float*)d_in[0];
    const float* Wf[6] = {(const float*)d_in[1], (const float*)d_in[2],
                          (const float*)d_in[3], (const float*)d_in[4],
                          (const float*)d_in[5], (const float*)d_in[6]};
    const float* cq   = (const float*)d_in[7];
    const float* ck   = (const float*)d_in[8];
    const float* cv   = (const float*)d_in[9];
    const float* Alb  = (const float*)d_in[10];
    const float* Ald  = (const float*)d_in[11];
    const float* dtb  = (const float*)d_in[12];
    const float* rmsw = (const float*)d_in[13];
    float* out = (float*)d_out;

    const int M = BATCH * SEQ_T;            // 8192
    const size_t P = (size_t)M * D_MODEL;   // 8,388,608 floats
    const size_t WSZ = (size_t)D_MODEL * D_MODEL;  // 1,048,576
    float* ws   = (float*)d_ws;
    float* qpre = ws + 0 * P;
    float* kpre = ws + 1 * P;
    float* vpre = ws + 2 * P;
    float* abuf = ws + 3 * P;               // a, then gk in place
    float* gbuf = ws + 4 * P;
    float* qb   = ws + 5 * P;
    float* kb   = ws + 6 * P;
    float* vb   = ws + 7 * P;
    float* obuf = ws + 8 * P;
    float* hc   = ws + 9 * P;               // 16 floats (64B-aligned region)
    __bf16* hsb = (__bf16*)(ws + 9 * P + 64);   // P bf16 (hs, later gated o)
    __bf16* wbb = hsb + P;                      // 6 x WSZ bf16 weights

    // --- bf16 preconversion (halves GEMM global traffic, removes hot-loop cvt)
    to_bf16_kernel<<<(int)(P / 8 / 256), 256, 0, stream>>>(hs, hsb, (long)(P / 8));
    for (int i = 0; i < 6; ++i)
        to_bf16_kernel<<<(int)(WSZ / 8 / 256), 256, 0, stream>>>(
            Wf[i], wbb + (size_t)i * WSZ, (long)(WSZ / 8));

    const dim3 ggrid(D_MODEL / 128, M / 128);   // (8, 64)
    gemm_bf16<<<ggrid, 256, 0, stream>>>(hsb, wbb + 0 * WSZ, qpre, M, D_MODEL, D_MODEL);
    gemm_bf16<<<ggrid, 256, 0, stream>>>(hsb, wbb + 1 * WSZ, kpre, M, D_MODEL, D_MODEL);
    gemm_bf16<<<ggrid, 256, 0, stream>>>(hsb, wbb + 2 * WSZ, vpre, M, D_MODEL, D_MODEL);
    gemm_bf16<<<ggrid, 256, 0, stream>>>(hsb, wbb + 3 * WSZ, abuf, M, D_MODEL, D_MODEL);
    gemm_bf16<<<ggrid, 256, 0, stream>>>(hsb, wbb + 4 * WSZ, gbuf, M, D_MODEL, D_MODEL);

    const int total4 = (int)(P / 4);        // 2,097,152
    conv_silu_kernel<<<total4 / 256, 256, 0, stream>>>(qpre, cq, qb, SEQ_T, total4);
    conv_silu_kernel<<<total4 / 256, 256, 0, stream>>>(kpre, ck, kb, SEQ_T, total4);
    conv_silu_kernel<<<total4 / 256, 256, 0, stream>>>(vpre, cv, vb, SEQ_T, total4);

    headconst_kernel<<<1, 32, 0, stream>>>(Alb, Ald, hc);
    gk_kernel<<<total4 / 256, 256, 0, stream>>>(abuf, dtb, hc, total4, SEQ_T);

    gla_kernel<<<BATCH * HEADS, 256, 0, stream>>>(qb, kb, vb, abuf, obuf, BATCH, SEQ_T);

    rmsgate_kernel<<<(M * HEADS) / 8, 256, 0, stream>>>(obuf, gbuf, rmsw, M * HEADS);

    // gated output -> bf16 (reuse hsb), then final projection
    to_bf16_kernel<<<(int)(P / 8 / 256), 256, 0, stream>>>(obuf, hsb, (long)(P / 8));
    gemm_bf16<<<ggrid, 256, 0, stream>>>(hsb, wbb + 5 * WSZ, out, M, D_MODEL, D_MODEL);
}
